// AffineCouplingLayer_1434519077071
// MI455X (gfx1250) — matrix operand order repaired and checked by reading, compiled, not run on hardware
//
#include <hip/hip_runtime.h>
#include <hip/hip_bf16.h>
#include <math.h>
#include <stdint.h>

typedef __bf16 bf16_t;
typedef __attribute__((ext_vector_type(16))) __bf16 v16bf;
typedef __attribute__((ext_vector_type(8)))  __bf16 v8bf;
typedef __attribute__((ext_vector_type(8)))  float  v8f;
typedef __attribute__((ext_vector_type(4)))  float  f4;
typedef __attribute__((ext_vector_type(4)))  unsigned int u32x4;
typedef __attribute__((ext_vector_type(8)))  int  i32x8;
typedef __attribute__((ext_vector_type(4)))  int  i32x4;

#define TM       128   // block tile M
#define TN       256   // block tile N
#define KSTEP    32
#define LDSPITCH 40    // bf16 elems per LDS row: 32 data + 8 pad (80B pitch, conflict-free)

#ifndef __has_builtin
#define __has_builtin(x) 0
#endif
#if __has_builtin(__builtin_amdgcn_tensor_load_to_lds) && __has_builtin(__builtin_amdgcn_s_wait_tensorcnt)
#define USE_TDM 1
#else
#define USE_TDM 0
#endif

// ---------------------------------------------------------------------------
// LDS fragment loaders per CDNA5 ISA VGPR layouts (wave32).
// A-matrix 16x32 bf16: lane<16 holds K {0..7,16..23}, lane>=16 holds {8..15,24..31}.
__device__ __forceinline__ v16bf frag_a(const bf16_t* tile, int lane) {
  const bf16_t* p = tile + (lane & 15) * LDSPITCH + ((lane >> 4) << 3);
  v8bf lo = *(const v8bf*)p;        // K kb .. kb+7
  v8bf hi = *(const v8bf*)(p + 16); // K kb+16 .. kb+23
  return __builtin_shufflevector(lo, hi, 0,1,2,3,4,5,6,7,8,9,10,11,12,13,14,15);
}
// B-matrix 32x16 bf16 (stored transposed, N x K in LDS): lane<16 holds K {0..7,16..23},
// lane>=16 holds K {8..15,24..31}, same as A.
__device__ __forceinline__ v16bf frag_b(const bf16_t* tile, int lane) {
  const bf16_t* p = tile + (lane & 15) * LDSPITCH + ((lane >> 4) << 3);
  v8bf lo = *(const v8bf*)p;
  v8bf hi = *(const v8bf*)(p + 16);
  return __builtin_shufflevector(lo, hi, 0,1,2,3,4,5,6,7,8,9,10,11,12,13,14,15);
}

#if USE_TDM
// Issue a TDM load of a (rows x 32) bf16 tile whose row stride in memory is
// row_stride_elems, into LDS at lds_byte_addr with 80B pitch (hardware padding:
// pad_interval=3 -> every 16 DWORDs, pad_amount=3 -> 4 DWORDs). D# bit layout
// per CDNA5 ISA; tensor_dim == tile_dim (no OOB).
// 6-arg builtin form (clang-23 / therock-10.0): (g0, g1, g2, g3, g4, cpol);
// trailing groups zero-filled (2D tensor -> VADDR2/3 disabled equivalent).
__device__ __forceinline__ void tdm_load_tile(const void* gptr, uint32_t lds_byte_addr,
                                              uint32_t rows, uint64_t row_stride_elems)
{
  const uint64_t ga = (uint64_t)(uintptr_t)gptr;
  u32x4 g0;
  g0[0] = 1u;                                                   // count=1 (valid), user mode
  g0[1] = lds_byte_addr;                                        // lds_addr
  g0[2] = (uint32_t)ga;                                         // global_addr[31:0]
  g0[3] = (uint32_t)((ga >> 32) & 0x01FFFFFFu) | 0x80000000u;   // global_addr[56:32] | type=2
  const uint32_t td0 = 32u, td1 = rows;
  i32x8 g1;
  g1[0] = (int)((1u << 16) | (1u << 20) | (3u << 22) | (3u << 25)); // data_size=2B, pad_en, intv=16DW, amt=4DW
  g1[1] = (int)((td0 & 0xFFFFu) << 16);                         // abar_addr=0 | tensor_dim0.lo
  g1[2] = (int)(((td0 >> 16) & 0xFFFFu) | ((td1 & 0xFFFFu) << 16));
  g1[3] = (int)(((td1 >> 16) & 0xFFFFu) | (32u << 16));         // tensor_dim1.hi | tile_dim0=32
  g1[4] = (int)(rows & 0xFFFFu);                                // tile_dim1 | tile_dim2=0
  g1[5] = (int)(uint32_t)(row_stride_elems & 0xFFFFFFFFu);      // tensor_dim0_stride.lo
  g1[6] = (int)(uint32_t)((row_stride_elems >> 32) & 0xFFFFu);  // stride.hi | stride1.lo=0
  g1[7] = 0;
  const i32x4 z4 = {0, 0, 0, 0};
  const i32x8 z8 = {0, 0, 0, 0, 0, 0, 0, 0};
  __builtin_amdgcn_tensor_load_to_lds(g0, g1, z4, z4, z8, 0);
}
#endif

// ---------------------------------------------------------------------------
// GEMM: C(MxN) = act(A(MxK,bf16) @ B + bias), B given transposed as Bt(NxK,bf16).
// Block tile 128x256, 256 threads (8 waves, 2x4 grid), wave tile 64x64 (4x4 WMMA).
template <bool LEAKY, bool OUT_BF16>
__global__ __launch_bounds__(256)
void gemm_bf16_wmma(const bf16_t* __restrict__ A,
                    const bf16_t* __restrict__ Bt,
                    const float*  __restrict__ bias,
                    void* __restrict__ Cout,
                    int M, int N, int K)
{
  __shared__ __align__(128) bf16_t sA[2][TM * LDSPITCH];
  __shared__ __align__(128) bf16_t sB[2][TN * LDSPITCH];

  const int tid  = threadIdx.x;
  const int lane = tid & 31;
  const int wid  = tid >> 5;
  const int wrow = wid >> 2;                 // 0..1 -> 64-row slab
  const int wcol = wid & 3;                  // 0..3 -> 64-col slab
  const int m0 = blockIdx.y * TM;
  const int n0 = blockIdx.x * TN;

  v8f acc[4][4];
  const v8f zero = {0.f,0.f,0.f,0.f,0.f,0.f,0.f,0.f};
#pragma unroll
  for (int i = 0; i < 4; ++i)
#pragma unroll
    for (int j = 0; j < 4; ++j) acc[i][j] = zero;

  const int KT = K / KSTEP;
  int buf = 0;

#if USE_TDM
  const bf16_t* Abase = A  + (size_t)m0 * K;
  const bf16_t* Bbase = Bt + (size_t)n0 * K;
  const uint32_t ldsA[2] = { (uint32_t)(uintptr_t)&sA[0][0], (uint32_t)(uintptr_t)&sA[1][0] };
  const uint32_t ldsB[2] = { (uint32_t)(uintptr_t)&sB[0][0], (uint32_t)(uintptr_t)&sB[1][0] };
  if (wid == 0) {                                   // one wave drives the TDM
    tdm_load_tile(Abase, ldsA[0], TM, (uint64_t)K);
    tdm_load_tile(Bbase, ldsB[0], TN, (uint64_t)K);
  }
#else
  // register-staged cooperative loads: 2 threads per row, 16 bf16 each
  const int lrow = tid >> 1;                 // 0..127
  const int lseg = (tid & 1) << 4;           // 0 or 16 (bf16 elems)
  const bf16_t* Ag  = A  + (size_t)(m0 + lrow) * K + lseg;
  const bf16_t* Bg0 = Bt + (size_t)(n0 + lrow) * K + lseg;
  const bf16_t* Bg1 = Bt + (size_t)(n0 + lrow + 128) * K + lseg;
  const int lofs  = lrow * LDSPITCH + lseg;
  const int lofs1 = (lrow + 128) * LDSPITCH + lseg;
  f4 a0 = *(const f4*)(Ag);       f4 a1 = *(const f4*)(Ag + 8);
  f4 b0 = *(const f4*)(Bg0);      f4 b1 = *(const f4*)(Bg0 + 8);
  f4 b2 = *(const f4*)(Bg1);      f4 b3 = *(const f4*)(Bg1 + 8);
  *(f4*)(&sA[0][lofs])      = a0; *(f4*)(&sA[0][lofs + 8])  = a1;
  *(f4*)(&sB[0][lofs])      = b0; *(f4*)(&sB[0][lofs + 8])  = b1;
  *(f4*)(&sB[0][lofs1])     = b2; *(f4*)(&sB[0][lofs1 + 8]) = b3;
#endif

  for (int kt = 0; kt < KT; ++kt) {
    __syncthreads();   // everyone done reading buf^1 from previous iteration
#if USE_TDM
    if (wid == 0) {
      if (kt + 1 < KT) {
        const int nb = buf ^ 1;
        tdm_load_tile(Abase + (size_t)(kt + 1) * KSTEP, ldsA[nb], TM, (uint64_t)K);
        tdm_load_tile(Bbase + (size_t)(kt + 1) * KSTEP, ldsB[nb], TN, (uint64_t)K);
        __builtin_amdgcn_s_wait_tensorcnt(2);   // tile kt's two DMAs complete
      } else {
        __builtin_amdgcn_s_wait_tensorcnt(0);
      }
    }
    __syncthreads();   // publish tile kt to all waves
#else
    const bool pref = (kt + 1 < KT);
    if (pref) {
      const bf16_t* Ap  = Ag  + (size_t)(kt + 1) * KSTEP;
      const bf16_t* Bp0 = Bg0 + (size_t)(kt + 1) * KSTEP;
      const bf16_t* Bp1 = Bg1 + (size_t)(kt + 1) * KSTEP;
      a0 = *(const f4*)(Ap);  a1 = *(const f4*)(Ap + 8);
      b0 = *(const f4*)(Bp0); b1 = *(const f4*)(Bp0 + 8);
      b2 = *(const f4*)(Bp1); b3 = *(const f4*)(Bp1 + 8);
      if (kt + 2 < KT) {
        __builtin_prefetch(Ag  + (size_t)(kt + 2) * KSTEP, 0, 0);
        __builtin_prefetch(Bg0 + (size_t)(kt + 2) * KSTEP, 0, 0);
        __builtin_prefetch(Bg1 + (size_t)(kt + 2) * KSTEP, 0, 0);
      }
    }
#endif

    const bf16_t* As = &sA[buf][0];
    const bf16_t* Bs = &sB[buf][0];
    v16bf af[4];
#pragma unroll
    for (int i = 0; i < 4; ++i)
      af[i] = frag_a(As + (wrow * 64 + i * 16) * LDSPITCH, lane);
#pragma unroll
    for (int j = 0; j < 4; ++j) {
      const v16bf bfr = frag_b(Bs + (wcol * 64 + j * 16) * LDSPITCH, lane);
#pragma unroll
      for (int i = 0; i < 4; ++i)
        acc[i][j] = __builtin_amdgcn_wmma_f32_16x16x32_bf16(
            false, af[i], false, bfr, (short)0, acc[i][j], false, false);
    }

#if !USE_TDM
    if (pref) {
      const int nb = buf ^ 1;
      *(f4*)(&sA[nb][lofs])      = a0; *(f4*)(&sA[nb][lofs + 8])  = a1;
      *(f4*)(&sB[nb][lofs])      = b0; *(f4*)(&sB[nb][lofs + 8])  = b1;
      *(f4*)(&sB[nb][lofs1])     = b2; *(f4*)(&sB[nb][lofs1 + 8]) = b3;
    }
#endif
    buf ^= 1;
  }

  // epilogue: bias + optional leaky-relu; C/D layout: lane l, elem v -> (M = v + 8*(l>=16), N = l&15)
  const int nlo  = lane & 15;
  const int mhal = (lane >> 4) << 3;
#pragma unroll
  for (int j = 0; j < 4; ++j) {
    const int ng = n0 + wcol * 64 + j * 16 + nlo;
    const float bv = bias[ng];
#pragma unroll
    for (int i = 0; i < 4; ++i) {
      const int mg = m0 + wrow * 64 + i * 16 + mhal;
#pragma unroll
      for (int v = 0; v < 8; ++v) {
        float c = acc[i][j][v] + bv;
        if (LEAKY) c = (c > 0.f) ? c : 0.2f * c;
        if (OUT_BF16) ((bf16_t*)Cout)[(size_t)(mg + v) * N + ng] = (bf16_t)c;
        else          ((float*) Cout)[(size_t)(mg + v) * N + ng] = c;
      }
    }
  }
}

// ---------------------------------------------------------------------------
__global__ void gather_bf16(const float* __restrict__ x, const int* __restrict__ idx,
                            bf16_t* __restrict__ Xm, int rows, int cols, int ld)
{
  size_t t = (size_t)blockIdx.x * blockDim.x + threadIdx.x;
  if (t >= (size_t)rows * cols) return;
  int r = (int)(t / cols), j = (int)(t % cols);
  Xm[t] = (bf16_t)x[(size_t)r * ld + idx[j]];
}

// W (K x N, f32 row-major) -> Wt (N x K, bf16 row-major)
__global__ void transpose_f32_bf16(const float* __restrict__ W, bf16_t* __restrict__ Wt,
                                   int K, int N)
{
  __shared__ float tile[32][33];
  const int n0 = blockIdx.x * 32, k0 = blockIdx.y * 32;
  const int tx = threadIdx.x, ty = threadIdx.y;
#pragma unroll
  for (int i = 0; i < 4; ++i)
    tile[ty + i * 8][tx] = W[(size_t)(k0 + ty + i * 8) * N + n0 + tx];
  __syncthreads();
#pragma unroll
  for (int i = 0; i < 4; ++i)
    Wt[(size_t)(n0 + ty + i * 8) * K + k0 + tx] = (bf16_t)tile[tx][ty + i * 8];
}

// per-row: s = 2*tanh(out[:, :Nh]); y_u = x_u*exp(s) + t; log_det = sum(s)
__global__ void coupling_epilogue(const float* __restrict__ x, const float* __restrict__ OUT,
                                  const int* __restrict__ idx_m, const int* __restrict__ idx_u,
                                  float* __restrict__ y, float* __restrict__ log_det,
                                  int L, int Nh)
{
  const int r   = blockIdx.x;
  const int tid = threadIdx.x;
  const float* xr   = x   + (size_t)r * L;
  const float* outr = OUT + (size_t)r * (2 * Nh);
  float* yr = y + (size_t)r * L;

  float acc = 0.f;
  for (int j = tid; j < Nh; j += blockDim.x) {
    const int cm = idx_m[j], cu = idx_u[j];
    yr[cm] = xr[cm];                          // masked columns pass through
    const float s = 2.f * tanhf(outr[j]);
    const float t = outr[Nh + j];
    yr[cu] = fmaf(xr[cu], expf(s), t);
    acc += s;
  }
  for (int o = 16; o > 0; o >>= 1) acc += __shfl_down(acc, o, 32);
  __shared__ float red[8];
  const int lane = tid & 31, wid = tid >> 5;
  if (lane == 0) red[wid] = acc;
  __syncthreads();
  if (tid == 0) {
    float s = 0.f;
    for (int i = 0; i < (int)(blockDim.x >> 5); ++i) s += red[i];
    log_det[r] = s;
  }
}

// ---------------------------------------------------------------------------
extern "C" void kernel_launch(void* const* d_in, const int* in_sizes, int n_in,
                              void* d_out, int out_size, void* d_ws, size_t ws_size,
                              hipStream_t stream)
{
  constexpr int Bn = 4096, L = 4096, Nh = 2048, H = 2048; // Nh = L/2
  const float* x   = (const float*)d_in[0];
  const float* W1  = (const float*)d_in[1];
  const float* b1  = (const float*)d_in[2];
  const float* W2  = (const float*)d_in[3];
  const float* b2  = (const float*)d_in[4];
  const float* W3  = (const float*)d_in[5];
  const float* b3  = (const float*)d_in[6];
  const int* idx_m = (const int*)d_in[7];
  const int* idx_u = (const int*)d_in[8];
  float* y       = (float*)d_out;
  float* log_det = y + (size_t)Bn * L;

  // workspace carve-up (~144 MiB total), 256B aligned
  char* wsb = (char*)d_ws;
  size_t off = 0;
  auto take = [&](size_t bytes) {
    char* p = wsb + off;
    off = (off + bytes + 255) & ~(size_t)255;
    return (void*)p;
  };
  bf16_t* Xm  = (bf16_t*)take((size_t)Bn * Nh * sizeof(bf16_t));
  bf16_t* W1t = (bf16_t*)take((size_t)H  * Nh * sizeof(bf16_t));      // H  x Nh
  bf16_t* W2t = (bf16_t*)take((size_t)H  * H  * sizeof(bf16_t));      // H  x H
  bf16_t* W3t = (bf16_t*)take((size_t)(2*Nh) * H * sizeof(bf16_t));   // 2Nh x H
  bf16_t* H1  = (bf16_t*)take((size_t)Bn * H * sizeof(bf16_t));
  bf16_t* H2  = (bf16_t*)take((size_t)Bn * H * sizeof(bf16_t));
  float*  OUTf= (float*) take((size_t)Bn * (2*Nh) * sizeof(float));
  (void)ws_size; (void)in_sizes; (void)n_in; (void)out_size;

  const dim3 blk(256);

  // prep: gather + convert x_m, transpose + convert weights to bf16 NxK
  {
    size_t total = (size_t)Bn * Nh;
    gather_bf16<<<dim3((unsigned)((total + 255) / 256)), blk, 0, stream>>>(x, idx_m, Xm, Bn, Nh, L);
  }
  const dim3 tb(32, 8);
  transpose_f32_bf16<<<dim3(H / 32,        Nh / 32), tb, 0, stream>>>(W1, W1t, Nh, H);
  transpose_f32_bf16<<<dim3(H / 32,        H  / 32), tb, 0, stream>>>(W2, W2t, H,  H);
  transpose_f32_bf16<<<dim3((2 * Nh) / 32, H  / 32), tb, 0, stream>>>(W3, W3t, H,  2 * Nh);

  // GEMM chain (WMMA bf16, f32 accumulate), block tile 128x256
  gemm_bf16_wmma<true,  true ><<<dim3(H / TN,        Bn / TM), blk, 0, stream>>>(
      Xm, W1t, b1, (void*)H1, Bn, H, Nh);
  gemm_bf16_wmma<true,  true ><<<dim3(H / TN,        Bn / TM), blk, 0, stream>>>(
      H1, W2t, b2, (void*)H2, Bn, H, H);
  gemm_bf16_wmma<false, false><<<dim3((2 * Nh) / TN, Bn / TM), blk, 0, stream>>>(
      H2, W3t, b3, (void*)OUTf, Bn, 2 * Nh, H);

  // fused tanh/exp/scatter + log-det reduction
  coupling_epilogue<<<dim3(Bn), blk, 0, stream>>>(x, OUTf, idx_m, idx_u, y, log_det, L, Nh);
}
